// MambaOp_45380624449606
// MI455X (gfx1250) — matrix operand order repaired and checked by reading, not compiled, not run on hardware
//
#include <hip/hip_runtime.h>

#define BB 4
#define LL 4096
#define DD 1024
#define BL (BB * LL)        // 16384 rows

#define BM 128
#define BN 128
#define BK 32
#define LDSS 40             // padded bf16 row stride in LDS (80B = 20 banks -> conflict-free)

#define TCH 128             // timesteps per scan chunk
#define NCH (LL / TCH)      // 32 chunks per (b,d)

typedef __bf16 bf16_t;
typedef __attribute__((ext_vector_type(16))) __bf16 v16bf;
typedef __attribute__((ext_vector_type(8)))  float  v8f;

union FragBF { unsigned int u[8]; v16bf v; };

__device__ __forceinline__ float sigmoidf(float x) {
    return 1.f / (1.f + __expf(-x));
}

// ---------------- f32 -> bf16 conversion (vectorized) ----------------
__global__ __launch_bounds__(256) void cvt_f32_to_bf16(
    const float* __restrict__ src, bf16_t* __restrict__ dst, int n) {
    int i = (blockIdx.x * 256 + threadIdx.x) * 4;
    if (i + 3 < n) {
        float4 f = *(const float4*)(src + i);
        union { bf16_t h[4]; uint2 u2; } pk;
        pk.h[0] = (bf16_t)f.x; pk.h[1] = (bf16_t)f.y;
        pk.h[2] = (bf16_t)f.z; pk.h[3] = (bf16_t)f.w;
        *(uint2*)(dst + i) = pk.u2;
    }
}

// ---------------- bf16 WMMA GEMM: C[M,N] = A[M,K] @ Bw[N,K]^T ----------------
// mode 0: N==2*DD, write out_u = C[:, :DD] (bf16), out_g = sigmoid(C[:, DD:]) (bf16)
// mode 1: out_f = C + resid (f32)
__global__ __launch_bounds__(256) void gemm_bf16_wmma(
    const bf16_t* __restrict__ A,    // M x K row-major
    const bf16_t* __restrict__ Bw,   // N x K row-major (weight (O,I))
    int M, int N, int K, int mode,
    bf16_t* __restrict__ out_u, bf16_t* __restrict__ out_g,
    const float* __restrict__ resid, float* __restrict__ out_f)
{
    __shared__ bf16_t lA[BM * LDSS];
    __shared__ bf16_t lB[BN * LDSS];

    const int tid  = threadIdx.x;
    const int lane = tid & 31;
    const int wave = tid >> 5;        // 8 waves
    const int wm   = wave & 3;        // 4 waves along M (32 rows each)
    const int wn   = wave >> 2;       // 2 waves along N (64 cols each)
    const int half = lane >> 4;
    const int l16  = lane & 15;

    const int m0 = blockIdx.x * BM;
    const int n0 = blockIdx.y * BN;

    const int lrow = tid >> 1;        // 0..127 : tile row loaded by this thread
    const int lcol = (tid & 1) * 16;  // 0 or 16

    v8f acc[2][4];
    #pragma unroll
    for (int i = 0; i < 2; ++i)
        #pragma unroll
        for (int j = 0; j < 4; ++j)
            #pragma unroll
            for (int e = 0; e < 8; ++e) acc[i][j][e] = 0.f;

    const bf16_t* gA = A  + (size_t)(m0 + lrow) * K + lcol;
    const bf16_t* gB = Bw + (size_t)(n0 + lrow) * K + lcol;

    for (int k0 = 0; k0 < K; k0 += BK) {
        float4 a0 = *(const float4*)(gA);
        float4 a1 = *(const float4*)(gA + 8);
        float4 b0 = *(const float4*)(gB);
        float4 b1 = *(const float4*)(gB + 8);
        if (k0 + BK < K) {            // pull next K-tile toward L0/L2
            __builtin_prefetch(gA + BK, 0, 3);
            __builtin_prefetch(gB + BK, 0, 3);
        }
        gA += BK; gB += BK;
        *(float4*)&lA[lrow * LDSS + lcol]     = a0;
        *(float4*)&lA[lrow * LDSS + lcol + 8] = a1;
        *(float4*)&lB[lrow * LDSS + lcol]     = b0;
        *(float4*)&lB[lrow * LDSS + lcol + 8] = b1;
        __syncthreads();

        // A fragments: lane holds row (M = l16), K packed per ISA 16-bit A layout:
        // VGPR p<4: K = 2p,2p+1 (+8 for lanes 16-31); p>=4: +16.
        FragBF fa[2], fb[4];
        #pragma unroll
        for (int sm = 0; sm < 2; ++sm) {
            const int row = wm * 32 + sm * 16 + l16;
            #pragma unroll
            for (int p = 0; p < 8; ++p) {
                const int kk = ((p >= 4) ? 16 : 0) + (p & 3) * 2 + 8 * half;
                fa[sm].u[p] = *(const unsigned int*)&lA[row * LDSS + kk];
            }
        }
        // B fragments: lane holds col (N = l16), same K packing as A: K = 8*half + 0..7, then 16 + 8*half + 0..7.
        #pragma unroll
        for (int sn = 0; sn < 4; ++sn) {
            const int col = wn * 64 + sn * 16 + l16;
            const bf16_t* p = &lB[col * LDSS + 8 * half];
            *(float4*)&fb[sn].u[0] = *(const float4*)p;
            *(float4*)&fb[sn].u[4] = *(const float4*)(p + 16);
        }
        #pragma unroll
        for (int sm = 0; sm < 2; ++sm)
            #pragma unroll
            for (int sn = 0; sn < 4; ++sn)
                acc[sm][sn] = __builtin_amdgcn_wmma_f32_16x16x32_bf16(
                    false, fa[sm].v, false, fb[sn].v,
                    (short)0, acc[sm][sn], false, false);
        __syncthreads();
    }

    // C layout per 16x16 tile: VGPR r -> row (r + 8*half), col l16
    if (mode == 0) {
        const int Nh = N >> 1;
        const bool is_g = (n0 >= Nh);     // whole 128-wide block is in one half
        #pragma unroll
        for (int sm = 0; sm < 2; ++sm)
            #pragma unroll
            for (int sn = 0; sn < 4; ++sn) {
                const int col = n0 + wn * 64 + sn * 16 + l16;
                const int cl  = is_g ? (col - Nh) : col;
                #pragma unroll
                for (int r = 0; r < 8; ++r) {
                    const int row = m0 + wm * 32 + sm * 16 + r + 8 * half;
                    const float v = acc[sm][sn][r];
                    if (is_g) out_g[(size_t)row * Nh + cl] = (bf16_t)sigmoidf(v);
                    else      out_u[(size_t)row * Nh + cl] = (bf16_t)v;
                }
            }
    } else {
        #pragma unroll
        for (int sm = 0; sm < 2; ++sm)
            #pragma unroll
            for (int sn = 0; sn < 4; ++sn) {
                const int col = n0 + wn * 64 + sn * 16 + l16;
                #pragma unroll
                for (int r = 0; r < 8; ++r) {
                    const int row = m0 + wm * 32 + sm * 16 + r + 8 * half;
                    const size_t o = (size_t)row * N + col;
                    out_f[o] = acc[sm][sn][r] + resid[o];
                }
            }
    }
}

// ---------------- chunked parallel scan over L ----------------
// Block: 256 threads = 64 channels x 4 chunks; grid (LL/(4*TCH), DD/64, BB)
__global__ __launch_bounds__(256) void scan_local(
    const bf16_t* __restrict__ xu, const float* __restrict__ conv_w,
    const float* __restrict__ decay_logit, const float* __restrict__ in_scale,
    float* __restrict__ carry)
{
    const int dl = threadIdx.x & 63;
    const int sub = threadIdx.x >> 6;
    const int d = blockIdx.y * 64 + dl;
    const int b = blockIdx.z;
    const int chunk = blockIdx.x * 4 + sub;       // 0..NCH-1
    const int t0 = chunk * TCH;

    const float w0 = conv_w[d * 3 + 0], w1 = conv_w[d * 3 + 1], w2 = conv_w[d * 3 + 2];
    const float decay = sigmoidf(decay_logit[d]);
    const float scale = (1.f - decay) * in_scale[d];

    const bf16_t* base = xu + (size_t)b * LL * DD + d;
    float xm2 = (t0 >= 2) ? (float)base[(size_t)(t0 - 2) * DD] : 0.f;
    float xm1 = (t0 >= 1) ? (float)base[(size_t)(t0 - 1) * DD] : 0.f;
    float s = 0.f;
    for (int i = 0; i < TCH; ++i) {
        const float xc = (float)base[(size_t)(t0 + i) * DD];
        const float u = scale * (w0 * xm2 + w1 * xm1 + w2 * xc);
        s = decay * s + u;
        xm2 = xm1; xm1 = xc;
    }
    carry[(((size_t)b * DD + d) << 5) + chunk] = s;
}

__global__ __launch_bounds__(256) void scan_carry_prefix(
    const float* __restrict__ decay_logit,
    const float* __restrict__ carry, float* __restrict__ prefix)
{
    const int idx = blockIdx.x * 256 + threadIdx.x; // (b*DD + d)
    const int d = idx & (DD - 1);
    const float decay = sigmoidf(decay_logit[d]);
    float dk = decay;
    #pragma unroll
    for (int i = 0; i < 7; ++i) dk *= dk;           // decay^128
    const size_t base = (size_t)idx << 5;
    float p = 0.f;
    for (int j = 0; j < NCH; ++j) {
        prefix[base + j] = p;
        p = dk * p + carry[base + j];
    }
}

__global__ __launch_bounds__(256) void scan_apply(
    const bf16_t* __restrict__ xu, const bf16_t* __restrict__ gbf,
    const float* __restrict__ conv_w, const float* __restrict__ decay_logit,
    const float* __restrict__ in_scale, const float* __restrict__ prefix,
    bf16_t* __restrict__ ybf)
{
    const int dl = threadIdx.x & 63;
    const int sub = threadIdx.x >> 6;
    const int d = blockIdx.y * 64 + dl;
    const int b = blockIdx.z;
    const int chunk = blockIdx.x * 4 + sub;
    const int t0 = chunk * TCH;

    const float w0 = conv_w[d * 3 + 0], w1 = conv_w[d * 3 + 1], w2 = conv_w[d * 3 + 2];
    const float decay = sigmoidf(decay_logit[d]);
    const float scale = (1.f - decay) * in_scale[d];

    const size_t chan = (size_t)b * LL * DD + d;
    const bf16_t* base = xu + chan;
    const bf16_t* gb = gbf + chan;
    bf16_t* yb = ybf + chan;

    float xm2 = (t0 >= 2) ? (float)base[(size_t)(t0 - 2) * DD] : 0.f;
    float xm1 = (t0 >= 1) ? (float)base[(size_t)(t0 - 1) * DD] : 0.f;
    float s = prefix[(((size_t)b * DD + d) << 5) + chunk];
    for (int i = 0; i < TCH; ++i) {
        const size_t off = (size_t)(t0 + i) * DD;
        const float xc = (float)base[off];
        const float u = scale * (w0 * xm2 + w1 * xm1 + w2 * xc);
        s = decay * s + u;
        xm2 = xm1; xm1 = xc;
        yb[off] = (bf16_t)(s * (float)gb[off]);
    }
}

// ---------------- in-place row RMSNorm on f32 output ----------------
__global__ __launch_bounds__(256) void rmsnorm_inplace(
    float* __restrict__ z, const float* __restrict__ norm_scale)
{
    const int row = blockIdx.x;
    float* p = z + (size_t)row * DD;
    float v[4]; float ss = 0.f;
    #pragma unroll
    for (int j = 0; j < 4; ++j) {
        v[j] = p[threadIdx.x + j * 256];
        ss += v[j] * v[j];
    }
    __shared__ float red[256];
    red[threadIdx.x] = ss;
    __syncthreads();
    for (int off = 128; off > 0; off >>= 1) {
        if (threadIdx.x < off) red[threadIdx.x] += red[threadIdx.x + off];
        __syncthreads();
    }
    const float rms = __fsqrt_rn(red[0] * (1.f / DD));
    const float inv = 1.f / (rms + 1e-8f);
    #pragma unroll
    for (int j = 0; j < 4; ++j) {
        const int c = threadIdx.x + j * 256;
        p[c] = norm_scale[c] * v[j] * inv;
    }
}

extern "C" void kernel_launch(void* const* d_in, const int* in_sizes, int n_in,
                              void* d_out, int out_size, void* d_ws, size_t ws_size,
                              hipStream_t stream) {
    (void)in_sizes; (void)n_in; (void)out_size; (void)ws_size;
    const float* x           = (const float*)d_in[0];
    const float* W_in        = (const float*)d_in[1];
    const float* conv_w      = (const float*)d_in[2];
    const float* decay_logit = (const float*)d_in[3];
    const float* in_scale    = (const float*)d_in[4];
    const float* W_out       = (const float*)d_in[5];
    const float* norm_scale  = (const float*)d_in[6];
    float* out = (float*)d_out;

    char* ws = (char*)d_ws;
    size_t off = 0;
    bf16_t* x_bf    = (bf16_t*)(ws + off); off += (size_t)BL * DD * 2;     // reused as y_bf
    bf16_t* Win_bf  = (bf16_t*)(ws + off); off += (size_t)2 * DD * DD * 2;
    bf16_t* Wout_bf = (bf16_t*)(ws + off); off += (size_t)DD * DD * 2;
    bf16_t* xu_bf   = (bf16_t*)(ws + off); off += (size_t)BL * DD * 2;
    bf16_t* g_bf    = (bf16_t*)(ws + off); off += (size_t)BL * DD * 2;
    float*  carry   = (float*)(ws + off);  off += (size_t)BB * DD * NCH * 4;
    float*  prefix  = (float*)(ws + off);  off += (size_t)BB * DD * NCH * 4;
    bf16_t* y_bf = x_bf;   // x_bf dead after GEMM1; reuse for gated scan output

    // 1) one-time bf16 conversion (x lives in L2 afterwards: 32 MB << 192 MB)
    cvt_f32_to_bf16<<<(BL * DD) / 1024, 256, 0, stream>>>(x, x_bf, BL * DD);
    cvt_f32_to_bf16<<<(2 * DD * DD) / 1024, 256, 0, stream>>>(W_in, Win_bf, 2 * DD * DD);
    cvt_f32_to_bf16<<<(DD * DD) / 1024, 256, 0, stream>>>(W_out, Wout_bf, DD * DD);

    // 2) GEMM1: x_proj = x @ W_in^T -> x_u (bf16), sigmoid(x_g) (bf16)
    gemm_bf16_wmma<<<dim3(BL / BM, (2 * DD) / BN), 256, 0, stream>>>(
        x_bf, Win_bf, BL, 2 * DD, DD, 0, xu_bf, g_bf, nullptr, nullptr);

    // 3) conv + drive + linear recurrence (chunked parallel scan) + gate
    dim3 sgrid(LL / (4 * TCH), DD / 64, BB);
    scan_local<<<sgrid, 256, 0, stream>>>(xu_bf, conv_w, decay_logit, in_scale, carry);
    scan_carry_prefix<<<(BB * DD) / 256, 256, 0, stream>>>(decay_logit, carry, prefix);
    scan_apply<<<sgrid, 256, 0, stream>>>(xu_bf, g_bf, conv_w, decay_logit,
                                          in_scale, prefix, y_bf);

    // 4) GEMM2: out = y @ W_out^T + x (f32)
    gemm_bf16_wmma<<<dim3(BL / BM, DD / BN), 256, 0, stream>>>(
        y_bf, Wout_bf, BL, DD, DD, 1, nullptr, nullptr, x, out);

    // 5) RMSNorm in place
    rmsnorm_inplace<<<BL, 256, 0, stream>>>(out, norm_scale);
}
